// PerformerAttention_69234872812311
// MI455X (gfx1250) — matrix operand order repaired and checked by reading, not compiled, not run on hardware
//
#include <hip/hip_runtime.h>
#include <hip/hip_bf16.h>
#include <math.h>

// Problem constants (match reference)
#define B_   4
#define N_   4096
#define DIM_ 1024
#define H_   16
#define DH_  64
#define M_   32

#define K_EPS      1e-4f
#define K_NRM      0.35355339059327373f   /* 64^-0.25 */
#define K_HALFNRM2 0.0625f                /* 0.5 * 64^-0.5 */
#define K_RATIO    0.17677669529663687f   /* 32^-0.5 */

// ---------------------------------------------------------------------------
// WMMA types (CDNA5 / gfx1250, wave32)
// ---------------------------------------------------------------------------
typedef __attribute__((ext_vector_type(16))) __bf16 v16bf;
typedef __attribute__((ext_vector_type(8)))  float  v8f;

union FragBF {
    unsigned int u[8];
    v16bf        v;
};

__device__ inline unsigned short f2bf(float f) {
    unsigned int u = __float_as_uint(f);
    unsigned int r = u + 0x7FFFu + ((u >> 16) & 1u);   // round-to-nearest-even
    return (unsigned short)(r >> 16);
}
__device__ inline unsigned int pack2bf(float a, float b) {
    return (unsigned int)f2bf(a) | ((unsigned int)f2bf(b) << 16);
}

__device__ inline void atomicMaxFloat(float* addr, float val) {
    if (val >= 0.0f) atomicMax((int*)addr, __float_as_int(val));
    else             atomicMin((unsigned int*)addr, __float_as_uint(val));
}

// ---------------------------------------------------------------------------
// Async global->LDS (per-lane, ASYNCcnt) and TDM (tensor DMA, TENSORcnt).
// ---------------------------------------------------------------------------
#if defined(__gfx1250__) && \
    __has_builtin(__builtin_amdgcn_global_load_async_to_lds_b128) && \
    __has_builtin(__builtin_amdgcn_s_wait_asynccnt)
#define USE_ASYNC_LDS 1
#else
#define USE_ASYNC_LDS 0
#endif

#if defined(__gfx1250__) && \
    __has_builtin(__builtin_amdgcn_tensor_load_to_lds) && \
    __has_builtin(__builtin_amdgcn_s_wait_tensorcnt)
#define USE_TDM 1
#else
#define USE_TDM 0
#endif

typedef __attribute__((__vector_size__(4 * sizeof(int))))  int          v4i_t;
typedef __attribute__((__vector_size__(4 * sizeof(int))))  unsigned int v4u_t;
typedef __attribute__((__vector_size__(8 * sizeof(int))))  int          v8i_t;

__device__ inline void async_copy16(const void* g, void* l) {
#if USE_ASYNC_LDS
    __builtin_amdgcn_global_load_async_to_lds_b128(
        (__attribute__((address_space(1))) v4i_t*)g,
        (__attribute__((address_space(3))) v4i_t*)l, 0, 0);
#else
    *(uint4*)l = *(const uint4*)g;
#endif
}
__device__ inline void wait_async() {
#if USE_ASYNC_LDS
    __builtin_amdgcn_s_wait_asynccnt(0);
#endif
}

// TDM: one descriptor moves a 128-row x 32-element (bf16) tile, row stride
// 1024 elements, from global into LDS.  D# bit layout per ISA ch.8.
// This toolchain declares the 6-arg builtin (extra int32x8 descriptor group).
__device__ inline void tdm_load_b_tile(const unsigned short* gtile, unsigned lds_byte) {
#if USE_TDM
    const unsigned long long ga = (unsigned long long)(size_t)gtile;
    const v4u_t g0 = {
        1u,                                           // count=1, user descriptor
        lds_byte,                                     // lds_addr [63:32]
        (unsigned)(ga & 0xFFFFFFFFu),                 // global_addr [95:64]
        (unsigned)((ga >> 32) & 0x01FFFFFFu) | 0x80000000u  // addr hi | type=2
    };
    const v8i_t g1 = {
        (int)0x00010000u,                             // data_size=2B, mask=0
        (int)((1024u & 0xFFFFu) << 16),               // tensor_dim0=1024 (lo16)
        (int)(((1024u >> 16) & 0xFFFFu)               // tensor_dim0 hi
              | ((131072u & 0xFFFFu) << 16)),         // tensor_dim1 lo16
        (int)(((131072u >> 16) & 0xFFFFu)             // tensor_dim1 hi
              | (32u << 16)),                         // tile_dim0=32 (K elems)
        (int)128,                                     // tile_dim1=128 rows
        (int)1024,                                    // tensor_dim0_stride lo32
        0, 0
    };
    const v4i_t gz4 = {0, 0, 0, 0};
    const v8i_t gz8 = {0, 0, 0, 0, 0, 0, 0, 0};
    __builtin_amdgcn_tensor_load_to_lds(g0, g1, gz4, gz4, gz8, 0);
#else
    (void)gtile; (void)lds_byte;
#endif
}
__device__ inline void wait_tensor() {
#if USE_TDM
    __builtin_amdgcn_s_wait_tensorcnt(0);
#endif
}

// ---------------------------------------------------------------------------
// Pre-pass A: fp32 -> bf16 elementwise.
// ---------------------------------------------------------------------------
__global__ __launch_bounds__(256) void performer_cvt_bf16(
    const float* __restrict__ src, unsigned short* __restrict__ dst)
{
    const size_t i = ((size_t)blockIdx.x * 256 + threadIdx.x) * 4;
    const float4 f = *(const float4*)(src + i);
    uint2 o;
    o.x = pack2bf(f.x, f.y);
    o.y = pack2bf(f.z, f.w);
    *(uint2*)(dst + i) = o;
}

// ---------------------------------------------------------------------------
// Pre-pass B: fp32 W[k][n] -> bf16 WT[n][k] (K-contiguous for B fragments).
// ---------------------------------------------------------------------------
__global__ __launch_bounds__(256) void performer_cvt_transpose_bf16(
    const float* __restrict__ W, unsigned short* __restrict__ WT)
{
    __shared__ float tile[32][33];
    const int t  = threadIdx.x;
    const int c0 = blockIdx.x * 32, r0 = blockIdx.y * 32;
    for (int i = t; i < 1024; i += 256) {
        int r = i >> 5, c = i & 31;
        tile[r][c] = W[(size_t)(r0 + r) * 1024 + c0 + c];
    }
    __syncthreads();
    for (int i = t; i < 1024; i += 256) {
        int c = i >> 5, r = i & 31;
        WT[(size_t)(c0 + c) * 1024 + r0 + r] = f2bf(tile[r][c]);
    }
}

// ---------------------------------------------------------------------------
// Kernel 1: C = A * W + bias.  A: bf16 [m][k] (direct global b128 fragment
// loads).  WT: bf16 [n][k].  B tiles double-buffered in LDS, staged by ONE
// tensor_load_to_lds DMA per K-step (wave 0 issues; s_wait_tensorcnt +
// barrier publishes); per-lane async-LDS fallback.  fp32 WMMA accumulate,
// 128x128 block tile, 8 waves x (16 rows x 8 col-tiles).
// ---------------------------------------------------------------------------
__global__ __launch_bounds__(256) void performer_gemm_bf16_wmma(
    const unsigned short* __restrict__ A, const unsigned short* __restrict__ WT,
    const float* __restrict__ bias, const unsigned char* __restrict__ rowmask,
    float* __restrict__ C)
{
    __shared__ unsigned short Bs[2][128 * 32];   // bf16 [n][k] tiles, 8KB each

    const int t    = threadIdx.x;
    const int lane = t & 31;
    const int wave = t >> 5;
    const int m0   = blockIdx.y * 128;
    const int n0   = blockIdx.x * 128;

    v8f acc[8];
#pragma unroll
    for (int i = 0; i < 8; ++i) acc[i] = v8f{};

    // stage buffer `buf` with the 128x32 WT tile at K-offset k0
    auto stage = [&](int buf, int k0) {
#if USE_TDM
        if (wave == 0) {
            tdm_load_b_tile(WT + (size_t)n0 * 1024 + k0,
                            (unsigned)(size_t)&Bs[buf][0]);
        }
#else
#pragma unroll
        for (int it = 0; it < 2; ++it) {
            const int c   = t + it * 256;       // 0..511 x 16B chunks
            const int col = c >> 2;
            const int seg = c & 3;
            async_copy16(WT + (size_t)(n0 + col) * 1024 + k0 + seg * 8,
                         &Bs[buf][col * 32 + seg * 8]);
        }
#endif
    };
    auto stage_wait = [&]() {
#if USE_TDM
        if (wave == 0) wait_tensor();
#else
        wait_async();
#endif
    };

    stage(0, 0);
    const int rowA = m0 + wave * 16 + (lane & 15);
    const int kb   = (lane < 16) ? 0 : 8;       // A frag: lanes 0-15 K 0-7/16-23

    for (int kt = 0; kt < 32; ++kt) {
        const int k0 = kt * 32;
        stage_wait();
        __syncthreads();                         // buffer kt&1 ready for all waves
        if (kt + 1 < 32) stage((kt + 1) & 1, k0 + 32);

        // ---- A fragment 16x32 bf16 straight from global (L2-resident) ----
        FragBF a;
        {
            const uint4 a0 = *(const uint4*)(A + (size_t)rowA * 1024 + k0 + kb);
            const uint4 a1 = *(const uint4*)(A + (size_t)rowA * 1024 + k0 + kb + 16);
            a.u[0] = a0.x; a.u[1] = a0.y; a.u[2] = a0.z; a.u[3] = a0.w;
            a.u[4] = a1.x; a.u[5] = a1.y; a.u[6] = a1.z; a.u[7] = a1.w;
        }
        __builtin_prefetch(A + (size_t)rowA * 1024 + k0 + 32, 0, 0);

        const unsigned int* Bs32 = (const unsigned int*)Bs[kt & 1];
        const int kwd = (lane < 16) ? 0 : 4;     // B frag: lanes 0-15 K 0-7/16-23
#pragma unroll
        for (int tn = 0; tn < 8; ++tn) {
            FragBF b;
            const int n = tn * 16 + (lane & 15);
#pragma unroll
            for (int j = 0; j < 8; ++j) b.u[j] = Bs32[n * 16 + kwd + (j & 3) + ((j >> 2) << 3)];
            acc[tn] = __builtin_amdgcn_wmma_f32_16x16x32_bf16(
                false, a.v, false, b.v, (short)0, acc[tn], false, false);
        }
    }

    // ---- epilogue: C/D layout — VGPR vi: M=vi (lanes 0-15) / 8+vi (16-31);
    //      N = lane%16 ----
    const int rbase = m0 + wave * 16 + ((lane >> 4) << 3);
#pragma unroll
    for (int tn = 0; tn < 8; ++tn) {
        const int col = n0 + tn * 16 + (lane & 15);
        const float bia = bias[col];
#pragma unroll
        for (int vi = 0; vi < 8; ++vi) {
            const int row = rbase + vi;
            float o = acc[tn][vi] + bia;
            if (rowmask && rowmask[row]) o = 0.0f;
            C[(size_t)row * 1024 + col] = o;
        }
    }
}

// ---------------------------------------------------------------------------
// Kernel 2: query feature map.  One wave per (b,h,n) row.  projT (fp32,
// pre-transposed [d][m] in workspace) is async-copied into LDS.
// ---------------------------------------------------------------------------
__global__ __launch_bounds__(256) void performer_feat_q(
    const float* __restrict__ q, const float* __restrict__ projT_ws,
    float* __restrict__ qf)
{
    __shared__ float projT[64 * 32];
    __shared__ float drow[8][64];
    const int t = threadIdx.x, lane = t & 31, w = t >> 5;

#pragma unroll
    for (int it = 0; it < 2; ++it) {
        const int c = t + it * 256;               // 512 x 16B chunks
        async_copy16(projT_ws + c * 4, &projT[c * 4]);
    }
    const int R = blockIdx.x * 8 + w;
    const int b = R / (H_ * N_);
    const int h = (R / N_) % H_;
    const int n = R % N_;
    const float* dp = q + ((size_t)(b * N_ + n)) * DIM_ + h * DH_;
    drow[w][lane]      = dp[lane];
    drow[w][lane + 32] = dp[lane + 32];
    wait_async();
    __syncthreads();

    float acc = 0.0f, x2 = 0.0f;
#pragma unroll 8
    for (int d = 0; d < 64; ++d) {
        float xv = drow[w][d];
        acc = fmaf(xv, projT[d * 32 + lane], acc);
        x2  = fmaf(xv, xv, x2);
    }
    const float dd   = K_NRM * acc;
    const float diag = x2 * K_HALFNRM2;
    float mx = dd;
#pragma unroll
    for (int o = 16; o >= 1; o >>= 1) mx = fmaxf(mx, __shfl_xor(mx, o, 32));
    qf[(size_t)R * M_ + lane] = K_RATIO * (expf(dd - diag - mx) + K_EPS);
}

// ---------------------------------------------------------------------------
// Kernel 3a: key feature pass 1 — store dd - diag, atomic-max dd per (b,h).
// ---------------------------------------------------------------------------
__global__ __launch_bounds__(256) void performer_feat_k_pass1(
    const float* __restrict__ k, const float* __restrict__ projT_ws,
    float* __restrict__ kfd, float* __restrict__ stabk)
{
    __shared__ float projT[64 * 32];
    __shared__ float drow[8][64];
    __shared__ float wmax[8];
    const int t = threadIdx.x, lane = t & 31, w = t >> 5;

#pragma unroll
    for (int it = 0; it < 2; ++it) {
        const int c = t + it * 256;
        async_copy16(projT_ws + c * 4, &projT[c * 4]);
    }
    const int R = blockIdx.x * 8 + w;
    const int b = R / (H_ * N_);
    const int h = (R / N_) % H_;
    const int n = R % N_;
    const float* dp = k + ((size_t)(b * N_ + n)) * DIM_ + h * DH_;
    drow[w][lane]      = dp[lane];
    drow[w][lane + 32] = dp[lane + 32];
    wait_async();
    __syncthreads();

    float acc = 0.0f, x2 = 0.0f;
#pragma unroll 8
    for (int d = 0; d < 64; ++d) {
        float xv = drow[w][d];
        acc = fmaf(xv, projT[d * 32 + lane], acc);
        x2  = fmaf(xv, xv, x2);
    }
    const float dd   = K_NRM * acc;
    const float diag = x2 * K_HALFNRM2;
    kfd[(size_t)R * M_ + lane] = dd - diag;

    float mx = dd;
#pragma unroll
    for (int o = 16; o >= 1; o >>= 1) mx = fmaxf(mx, __shfl_xor(mx, o, 32));
    if (lane == 0) wmax[w] = mx;
    __syncthreads();
    if (t == 0) {
        float bm = wmax[0];
#pragma unroll
        for (int i = 1; i < 8; ++i) bm = fmaxf(bm, wmax[i]);
        atomicMaxFloat(&stabk[b * H_ + h], bm);
    }
}

// ---------------------------------------------------------------------------
// Kernel 3b: finalize kf in place + deterministic ksum (fixed-order reduce).
// ---------------------------------------------------------------------------
__global__ __launch_bounds__(256) void performer_feat_k_pass2(
    float* __restrict__ kfd, const float* __restrict__ stabk,
    float* __restrict__ ksum)
{
    __shared__ float psum[8][M_];
    const int t = threadIdx.x;
    const int m = t & 31, s = t >> 5;
    const int bh = blockIdx.x;
    const float stab = stabk[bh];
    const size_t base = (size_t)bh * N_ * M_;

    float local = 0.0f;
    for (int i = 0; i < N_ / 8; ++i) {
        const int n = i * 8 + s;
        const size_t idx = base + (size_t)n * M_ + m;
        const float val = K_RATIO * (expf(kfd[idx] - stab) + K_EPS);
        kfd[idx] = val;
        local += val;
    }
    psum[s][m] = local;
    __syncthreads();
    if (t < M_) {
        float acc2 = 0.0f;
#pragma unroll
        for (int s2 = 0; s2 < 8; ++s2) acc2 += psum[s2][t];
        ksum[bh * M_ + t] = acc2;
    }
}

// ---------------------------------------------------------------------------
// Kernel 4: context[b,h,m,e] = sum_n kf[b,h,n,m] * v[b,n,h*64+e].
// ---------------------------------------------------------------------------
__global__ __launch_bounds__(256) void performer_context(
    const float* __restrict__ kf, const float* __restrict__ v,
    float* __restrict__ ctx)
{
    const int bh = blockIdx.x;
    const int b = bh / H_, h = bh % H_;
    const int t = threadIdx.x;
    const int m = t & 31;
    const int eb = (t >> 5) * 8;

    const float* vb = v + (size_t)b * N_ * DIM_ + h * DH_ + eb;
    const float* kb = kf + (size_t)bh * N_ * M_ + m;

    float acc[8];
#pragma unroll
    for (int j = 0; j < 8; ++j) acc[j] = 0.0f;

    for (int n = 0; n < N_; ++n) {
        const float kv = kb[(size_t)n * M_];
        const float4 v0 = *(const float4*)(vb + (size_t)n * DIM_);
        const float4 v1 = *(const float4*)(vb + (size_t)n * DIM_ + 4);
        acc[0] = fmaf(kv, v0.x, acc[0]); acc[1] = fmaf(kv, v0.y, acc[1]);
        acc[2] = fmaf(kv, v0.z, acc[2]); acc[3] = fmaf(kv, v0.w, acc[3]);
        acc[4] = fmaf(kv, v1.x, acc[4]); acc[5] = fmaf(kv, v1.y, acc[5]);
        acc[6] = fmaf(kv, v1.z, acc[6]); acc[7] = fmaf(kv, v1.w, acc[7]);
    }
    float* cp = ctx + (size_t)bh * M_ * DH_ + m * DH_ + eb;
#pragma unroll
    for (int j = 0; j < 8; ++j) cp[j] = acc[j];
}

// ---------------------------------------------------------------------------
// Kernel 5: attn_bf16[b,n,h*64+e] = Dinv * sum_m qf[n,m]*ctx[m,e].
// ctx tile (8KB) async-copied into LDS.
// ---------------------------------------------------------------------------
__global__ __launch_bounds__(256) void performer_out(
    const float* __restrict__ qf, const float* __restrict__ ctx,
    const float* __restrict__ ksum, unsigned short* __restrict__ attn)
{
    __shared__ float ctxs[M_ * DH_];
    __shared__ float ks[M_];
    __shared__ float qrow[8][M_];
    const int t = threadIdx.x, lane = t & 31, w = t >> 5;
    const int bh = blockIdx.x / (N_ / 8);
    const int nc = blockIdx.x % (N_ / 8);
    const int b = bh / H_, h = bh % H_;
    const int n = nc * 8 + w;

#pragma unroll
    for (int it = 0; it < 2; ++it) {
        const int c = t + it * 256;               // 512 x 16B chunks
        async_copy16(ctx + (size_t)bh * M_ * DH_ + c * 4, &ctxs[c * 4]);
    }
    if (t < M_) ks[t] = ksum[bh * M_ + t];
    qrow[w][lane] = qf[((size_t)bh * N_ + n) * M_ + lane];
    wait_async();
    __syncthreads();

    float a0 = 0.0f, a1 = 0.0f, D = 0.0f;
#pragma unroll 4
    for (int m = 0; m < M_; ++m) {
        const float qv = qrow[w][m];
        D  = fmaf(qv, ks[m], D);
        a0 = fmaf(qv, ctxs[m * DH_ + lane], a0);
        a1 = fmaf(qv, ctxs[m * DH_ + lane + 32], a1);
    }
    const float Dinv = 1.0f / D;
    unsigned short* op = attn + ((size_t)(b * N_ + n)) * DIM_ + h * DH_;
    op[lane]      = f2bf(a0 * Dinv);
    op[lane + 32] = f2bf(a1 * Dinv);
}

// ---------------------------------------------------------------------------
// Init: stab_k = -inf; build projT_ws (fp32, [d][m]) once.
// ---------------------------------------------------------------------------
__global__ __launch_bounds__(256) void performer_init(
    const float* __restrict__ proj, float* __restrict__ projT_ws,
    float* __restrict__ stabk)
{
    const int t = threadIdx.x;
    if (t < B_ * H_) stabk[t] = -INFINITY;
    for (int i = t; i < M_ * DH_; i += 256) {
        const int m = i >> 6, d = i & 63;         // proj is (M,DH) row-major
        projT_ws[d * 32 + m] = proj[i];
    }
}

// ---------------------------------------------------------------------------
extern "C" void kernel_launch(void* const* d_in, const int* in_sizes, int n_in,
                              void* d_out, int out_size, void* d_ws, size_t ws_size,
                              hipStream_t stream) {
    (void)in_sizes; (void)n_in; (void)out_size; (void)ws_size;
    const float*         x    = (const float*)d_in[0];
    const unsigned char* mask = (const unsigned char*)d_in[1];
    const float*         proj = (const float*)d_in[2];
    const float* Wq = (const float*)d_in[3]; const float* bq = (const float*)d_in[4];
    const float* Wk = (const float*)d_in[5]; const float* bk = (const float*)d_in[6];
    const float* Wv = (const float*)d_in[7]; const float* bv = (const float*)d_in[8];
    const float* Wo = (const float*)d_in[9]; const float* bo = (const float*)d_in[10];
    float* out = (float*)d_out;

    char* p = (char*)d_ws;
    auto carve = [&](size_t bytes) { char* r = p; p += (bytes + 255) & ~(size_t)255; return r; };
    const size_t ND = (size_t)B_ * N_ * DIM_;         // 16.7M elements
    const size_t NF = (size_t)B_ * H_ * N_ * M_;      // 8.4M elements

    float*          q      = (float*)carve(ND * 4);
    float*          k      = (float*)carve(ND * 4);
    float*          v      = (float*)carve(ND * 4);
    float*          qf     = (float*)carve(NF * 4);
    float*          kfd    = (float*)carve(NF * 4);
    float*          ctx    = (float*)carve((size_t)B_ * H_ * M_ * DH_ * 4);
    float*          ksum   = (float*)carve((size_t)B_ * H_ * M_ * 4);
    float*          stabk  = (float*)carve((size_t)B_ * H_ * 4);
    float*          projT  = (float*)carve((size_t)M_ * DH_ * 4);
    unsigned short* xbf    = (unsigned short*)carve(ND * 2);
    unsigned short* attnbf = (unsigned short*)carve(ND * 2);
    unsigned short* WqT    = (unsigned short*)carve((size_t)DIM_ * DIM_ * 2);
    unsigned short* WkT    = (unsigned short*)carve((size_t)DIM_ * DIM_ * 2);
    unsigned short* WvT    = (unsigned short*)carve((size_t)DIM_ * DIM_ * 2);
    unsigned short* WoT    = (unsigned short*)carve((size_t)DIM_ * DIM_ * 2);

    const dim3 blk(256);
    const dim3 ggrid(DIM_ / 128, (B_ * N_) / 128);    // (8, 128)
    const dim3 tgrid(DIM_ / 32, DIM_ / 32);           // (32, 32)

    // One-shot precision/layout conversion (x stays L2-resident as bf16)
    performer_cvt_bf16<<<ND / 1024, blk, 0, stream>>>(x, xbf);
    performer_cvt_transpose_bf16<<<tgrid, blk, 0, stream>>>(Wq, WqT);
    performer_cvt_transpose_bf16<<<tgrid, blk, 0, stream>>>(Wk, WkT);
    performer_cvt_transpose_bf16<<<tgrid, blk, 0, stream>>>(Wv, WvT);
    performer_cvt_transpose_bf16<<<tgrid, blk, 0, stream>>>(Wo, WoT);

    // QKV projections (WMMA bf16, TDM-staged double buffering)
    performer_gemm_bf16_wmma<<<ggrid, blk, 0, stream>>>(xbf, WqT, bq, nullptr, q);
    performer_gemm_bf16_wmma<<<ggrid, blk, 0, stream>>>(xbf, WkT, bk, nullptr, k);
    performer_gemm_bf16_wmma<<<ggrid, blk, 0, stream>>>(xbf, WvT, bv, mask,    v);

    performer_init<<<1, 256, 0, stream>>>(proj, projT, stabk);

    // Feature maps (fp32 VALU; exp is precision-sensitive, FLOPs negligible)
    performer_feat_q<<<B_ * H_ * N_ / 8, blk, 0, stream>>>(q, projT, qf);
    performer_feat_k_pass1<<<B_ * H_ * N_ / 8, blk, 0, stream>>>(k, projT, kfd, stabk);
    performer_feat_k_pass2<<<B_ * H_, blk, 0, stream>>>(kfd, stabk, ksum);

    // Linear attention
    performer_context<<<B_ * H_, blk, 0, stream>>>(kfd, v, ctx);
    performer_out<<<B_ * H_ * N_ / 8, blk, 0, stream>>>(qf, ctx, ksum, attnbf);

    // Output projection (WMMA bf16)
    performer_gemm_bf16_wmma<<<ggrid, blk, 0, stream>>>(attnbf, WoT, bo, nullptr, out);
}
